// GRUModel_45414984188111
// MI455X (gfx1250) — matrix operand order repaired and checked by reading, not compiled, not run on hardware
//
#include <hip/hip_runtime.h>
#include <hip/hip_bf16.h>

// ---------------- CDNA5 (gfx1250) wave32 WMMA types ----------------
typedef __attribute__((ext_vector_type(16))) _Float16 v16h;
typedef __attribute__((ext_vector_type(8)))  _Float16 v8h;
typedef __attribute__((ext_vector_type(8)))  float    v8f;

#define NB     8192   // B*ENC rows
#define DM     512    // d_model
#define D3     1536   // 3*d_model
#define SEGL   64     // segment length
#define SNX    16     // encoder segments
#define SNY    8      // decoder segments
#define ENCN   64
#define SEQN   1024
#define PREDN  512
#define RD     65536  // B*ENC*SNY decoder rows

#if __has_builtin(__builtin_amdgcn_rcpf)
#define RCPF(x) __builtin_amdgcn_rcpf(x)
#else
#define RCPF(x) (1.0f / (x))
#endif

// D = A(16xK) * B(Kx16) + C, K step 32, f16 in / f32 acc
__device__ __forceinline__ v8f wmma16(v16h a, v16h b, v8f c) {
  return __builtin_amdgcn_wmma_f32_16x16x32_f16(false, a, false, b, (short)0, c,
                                                false, false);
}

// A-fragment: row pointer for this lane's M-row (m = lane&15), k-block k0.
// A 16x32 layout (matches B): lanes 0-15 hold K {0..15}, lanes 16-31 hold
// K {16..31} -> two contiguous 16B loads.
__device__ __forceinline__ v16h load_a_row(const _Float16* __restrict__ row,
                                           int k0, int lane) {
  int c0 = k0 + ((lane & 16) ? 16 : 0);
  v8h lo = *(const v8h*)(row + c0);
  v8h hi = *(const v8h*)(row + c0 + 8);
  v16h a;
#pragma unroll
  for (int i = 0; i < 8; ++i) { a[i] = lo[i]; a[i + 8] = hi[i]; }
  return a;
}

// B-fragment for C = X @ W^T with W row-major [ncols x K]:
// B[k][n] = W[n][k]; lane's column n = lane&15, K kb..kb+15 contiguous (32B).
__device__ __forceinline__ v16h load_b_frag(const _Float16* __restrict__ W,
                                            int ldw, int k0, int lane) {
  int n  = lane & 15;
  int kb = k0 + ((lane & 16) ? 16 : 0);
  return *(const v16h*)(W + n * ldw + kb);
}

// Fast activations: v_exp_f32 + v_rcp_f32 only (no IEEE div expansion).
__device__ __forceinline__ float sigf(float x) {
  return RCPF(1.0f + __expf(-x));
}
__device__ __forceinline__ float tanh_fast(float x) {
  return 2.0f * RCPF(1.0f + __expf(-2.0f * x)) - 1.0f;
}

// ---------------- elementwise / prep kernels ----------------
__global__ __launch_bounds__(256) void k_convert(const float* __restrict__ in,
                                                 _Float16* __restrict__ out, int n) {
  int i = blockIdx.x * 256 + threadIdx.x;
  if (i < n) out[i] = (_Float16)in[i];
}

__global__ __launch_bounds__(256) void k_zero_h(float* __restrict__ h32,
                                                _Float16* __restrict__ h16) {
  int i = blockIdx.x * 256 + threadIdx.x;  // NB*DM threads
  h32[i] = 0.0f;
  h16[i] = (_Float16)0.0f;
}

// xt[r*64+s] = x[b, t*64+s, c] - x[b, 1023, c],  r = b*64 + c
__global__ __launch_bounds__(256) void k_xt(const float* __restrict__ x,
                                            _Float16* __restrict__ xt, int t) {
  int i = blockIdx.x * 256 + threadIdx.x;  // NB*SEGL threads
  int r = i >> 6, s = i & 63;
  int b = r >> 6, c = r & 63;
  float last = x[(b * SEQN + (SEQN - 1)) * ENCN + c];
  float v    = x[(b * SEQN + t * SEGL + s) * ENCN + c] - last;
  xt[i] = (_Float16)v;
}

// pe table (512 rows x 512): [pos_emb(p) | channel_emb(c)], row q = c*8+p
__global__ __launch_bounds__(256) void k_pe(const float* __restrict__ pos,
                                            const float* __restrict__ chan,
                                            _Float16* __restrict__ pe) {
  int i = blockIdx.x * 256 + threadIdx.x;  // 512*DM threads
  int q = i >> 9, j = i & 511;
  int c = q >> 3, p = q & 7;
  float v = (j < 256) ? pos[p * 256 + j] : chan[c * 256 + (j - 256)];
  pe[i] = (_Float16)v;
}

// ---------------- WMMA GEMM kernels ----------------
// Block mapping convention: the 8 waves of a block share one n0 (same B
// fragments -> L0 reuse) and take 8 consecutive M tiles.

// emb = silu(xt @ W_emb^T + b_emb): M tiles = NB/16, N tiles = DM/16.
__global__ __launch_bounds__(256) void k_emb(const _Float16* __restrict__ xt,
                                             const _Float16* __restrict__ Wemb,
                                             const float* __restrict__ bemb,
                                             float* __restrict__ e32,
                                             _Float16* __restrict__ e16) {
  int wave = threadIdx.x >> 5, lane = threadIdx.x & 31;
  int n0 = (blockIdx.x & 31) << 4;
  int m0 = (((blockIdx.x >> 5) << 3) + wave) << 4;
  const _Float16* arow = xt + (m0 + (lane & 15)) * SEGL;
  const _Float16* Wb   = Wemb + n0 * SEGL;
  v8f acc = {};
#pragma unroll
  for (int k0 = 0; k0 < SEGL; k0 += 32) {
    acc = wmma16(load_a_row(arow, k0, lane), load_b_frag(Wb, SEGL, k0, lane), acc);
  }
  int n  = n0 + (lane & 15);
  int mb = m0 + ((lane & 16) ? 8 : 0);
  float bias = bemb[n];
#pragma unroll
  for (int v = 0; v < 8; ++v) {
    float t = acc[v] + bias;
    float s = t * sigf(t);  // SiLU
    int idx = (mb + v) * DM + n;
    e32[idx] = s;
    e16[idx] = (_Float16)s;
  }
}

// Fused GRU-cell gates (encoder), M=32 per wave:
// 12 accumulators; every B fragment feeds two WMMAs.
__global__ __launch_bounds__(256) void k_gates(const _Float16* __restrict__ X16,
                                               const _Float16* __restrict__ H16,
                                               const float* __restrict__ H32,
                                               const _Float16* __restrict__ Wih,
                                               const _Float16* __restrict__ Whh,
                                               const float* __restrict__ bih,
                                               const float* __restrict__ bhh,
                                               _Float16* __restrict__ hcell) {
  int wave = threadIdx.x >> 5, lane = threadIdx.x & 31;
  int n0 = (blockIdx.x & 31) << 4;
  int m0 = ((((blockIdx.x >> 5) << 3) + wave) << 5);  // 32 rows per wave
  const _Float16* ax0 = X16 + (m0 + (lane & 15)) * DM;
  const _Float16* ax1 = ax0 + 16 * DM;
  const _Float16* ah0 = H16 + (m0 + (lane & 15)) * DM;
  const _Float16* ah1 = ah0 + 16 * DM;
  const _Float16* Bw = Wih + n0 * DM;   // + g*DM*DM for gate g
  const _Float16* Cw = Whh + n0 * DM;
  v8f xr0 = {}, xz0 = {}, xn0 = {}, hr0 = {}, hz0 = {}, hn0 = {};
  v8f xr1 = {}, xz1 = {}, xn1 = {}, hr1 = {}, hz1 = {}, hn1 = {};
  for (int k0 = 0; k0 < DM; k0 += 32) {
    v16h a0 = load_a_row(ax0, k0, lane);
    v16h a1 = load_a_row(ax1, k0, lane);
    v16h b  = load_b_frag(Bw, DM, k0, lane);
    xr0 = wmma16(a0, b, xr0); xr1 = wmma16(a1, b, xr1);
    b = load_b_frag(Bw + DM * DM, DM, k0, lane);
    xz0 = wmma16(a0, b, xz0); xz1 = wmma16(a1, b, xz1);
    b = load_b_frag(Bw + 2 * DM * DM, DM, k0, lane);
    xn0 = wmma16(a0, b, xn0); xn1 = wmma16(a1, b, xn1);
    a0 = load_a_row(ah0, k0, lane);
    a1 = load_a_row(ah1, k0, lane);
    b = load_b_frag(Cw, DM, k0, lane);
    hr0 = wmma16(a0, b, hr0); hr1 = wmma16(a1, b, hr1);
    b = load_b_frag(Cw + DM * DM, DM, k0, lane);
    hz0 = wmma16(a0, b, hz0); hz1 = wmma16(a1, b, hz1);
    b = load_b_frag(Cw + 2 * DM * DM, DM, k0, lane);
    hn0 = wmma16(a0, b, hn0); hn1 = wmma16(a1, b, hn1);
  }
  int n = n0 + (lane & 15);
  float bri = bih[n], bzi = bih[DM + n], bni = bih[2 * DM + n];
  float brh = bhh[n], bzh = bhh[DM + n], bnh = bhh[2 * DM + n];
  auto epi = [&](v8f xr, v8f xz, v8f xn, v8f hr, v8f hz, v8f hn, int mb) {
#pragma unroll
    for (int v = 0; v < 8; ++v) {
      float rg = sigf(xr[v] + bri + hr[v] + brh);
      float zg = sigf(xz[v] + bzi + hz[v] + bzh);
      float ng = tanh_fast(xn[v] + bni + rg * (hn[v] + bnh));
      int idx = (mb + v) * DM + n;
      float hp = H32[idx];
      hcell[idx] = (_Float16)((1.0f - zg) * ng + zg * hp);
    }
  };
  int off = (lane & 16) ? 8 : 0;
  epi(xr0, xz0, xn0, hr0, hz0, hn0, m0 + off);
  epi(xr1, xz1, xn1, hr1, hz1, hn1, m0 + 16 + off);
}

// h_new = emb + h_cell @ res_W^T + res_b  (writes f32 + f16 mirror)
__global__ __launch_bounds__(256) void k_res(const _Float16* __restrict__ hc,
                                             const _Float16* __restrict__ Wr,
                                             const float* __restrict__ rb,
                                             const float* __restrict__ e32,
                                             float* __restrict__ h32,
                                             _Float16* __restrict__ h16) {
  int wave = threadIdx.x >> 5, lane = threadIdx.x & 31;
  int n0 = (blockIdx.x & 31) << 4;
  int m0 = (((blockIdx.x >> 5) << 3) + wave) << 4;
  const _Float16* arow = hc + (m0 + (lane & 15)) * DM;
  const _Float16* Wb   = Wr + n0 * DM;
  v8f acc = {};
  for (int k0 = 0; k0 < DM; k0 += 32) {
    acc = wmma16(load_a_row(arow, k0, lane), load_b_frag(Wb, DM, k0, lane), acc);
  }
  int n  = n0 + (lane & 15);
  int mb = m0 + ((lane & 16) ? 8 : 0);
  float bias = rb[n];
#pragma unroll
  for (int v = 0; v < 8; ++v) {
    int idx = (mb + v) * DM + n;
    float o = e32[idx] + acc[v] + bias;
    h32[idx] = o;
    h16[idx] = (_Float16)o;
  }
}

// Decoder fused gates, M=32 per wave. Row r: pe row = r&511, hidden row = r>>3.
__global__ __launch_bounds__(256) void k_dec(const _Float16* __restrict__ pe,
                                             const _Float16* __restrict__ hn16,
                                             const float* __restrict__ hn32,
                                             const _Float16* __restrict__ Wih,
                                             const _Float16* __restrict__ Whh,
                                             const float* __restrict__ bih,
                                             const float* __restrict__ bhh,
                                             _Float16* __restrict__ hy) {
  int wave = threadIdx.x >> 5, lane = threadIdx.x & 31;
  int n0 = (blockIdx.x & 31) << 4;
  int m0 = ((((blockIdx.x >> 5) << 3) + wave) << 5);  // 32 rows per wave
  const _Float16* ax0 = pe + ((m0 & 511) + (lane & 15)) * DM;
  const _Float16* ax1 = pe + (((m0 + 16) & 511) + (lane & 15)) * DM;
  const _Float16* ah0 = hn16 + ((m0 + (lane & 15)) >> 3) * DM;
  const _Float16* ah1 = hn16 + ((m0 + 16 + (lane & 15)) >> 3) * DM;
  const _Float16* Bw = Wih + n0 * DM;
  const _Float16* Cw = Whh + n0 * DM;
  v8f xr0 = {}, xz0 = {}, xn0 = {}, hr0 = {}, hz0 = {}, hn0 = {};
  v8f xr1 = {}, xz1 = {}, xn1 = {}, hr1 = {}, hz1 = {}, hn1 = {};
  for (int k0 = 0; k0 < DM; k0 += 32) {
    v16h a0 = load_a_row(ax0, k0, lane);
    v16h a1 = load_a_row(ax1, k0, lane);
    v16h b  = load_b_frag(Bw, DM, k0, lane);
    xr0 = wmma16(a0, b, xr0); xr1 = wmma16(a1, b, xr1);
    b = load_b_frag(Bw + DM * DM, DM, k0, lane);
    xz0 = wmma16(a0, b, xz0); xz1 = wmma16(a1, b, xz1);
    b = load_b_frag(Bw + 2 * DM * DM, DM, k0, lane);
    xn0 = wmma16(a0, b, xn0); xn1 = wmma16(a1, b, xn1);
    a0 = load_a_row(ah0, k0, lane);
    a1 = load_a_row(ah1, k0, lane);
    b = load_b_frag(Cw, DM, k0, lane);
    hr0 = wmma16(a0, b, hr0); hr1 = wmma16(a1, b, hr1);
    b = load_b_frag(Cw + DM * DM, DM, k0, lane);
    hz0 = wmma16(a0, b, hz0); hz1 = wmma16(a1, b, hz1);
    b = load_b_frag(Cw + 2 * DM * DM, DM, k0, lane);
    hn0 = wmma16(a0, b, hn0); hn1 = wmma16(a1, b, hn1);
  }
  int n = n0 + (lane & 15);
  float bri = bih[n], bzi = bih[DM + n], bni = bih[2 * DM + n];
  float brh = bhh[n], bzh = bhh[DM + n], bnh = bhh[2 * DM + n];
  auto epi = [&](v8f xr, v8f xz, v8f xn, v8f hr, v8f hz, v8f hn, int mb) {
#pragma unroll
    for (int v = 0; v < 8; ++v) {
      int r = mb + v;
      float rg = sigf(xr[v] + bri + hr[v] + brh);
      float zg = sigf(xz[v] + bzi + hz[v] + bzh);
      float ng = tanh_fast(xn[v] + bni + rg * (hn[v] + bnh));
      float hp = hn32[(r >> 3) * DM + n];
      hy[r * DM + n] = (_Float16)((1.0f - zg) * ng + zg * hp);
    }
  };
  int off = (lane & 16) ? 8 : 0;
  epi(xr0, xz0, xn0, hr0, hz0, hn0, m0 + off);
  epi(xr1, xz1, xn1, hr1, hz1, hn1, m0 + 16 + off);
}

// y = hy @ pred_W^T + pred_b, scattered to (B, PRED, ENC) with seq_last added.
__global__ __launch_bounds__(256) void k_pred(const _Float16* __restrict__ hy,
                                              const _Float16* __restrict__ Wp,
                                              const float* __restrict__ pb,
                                              const float* __restrict__ x,
                                              float* __restrict__ out) {
  int wave = threadIdx.x >> 5, lane = threadIdx.x & 31;
  int s0 = (blockIdx.x & 3) << 4;
  int m0 = (((blockIdx.x >> 2) << 3) + wave) << 4;
  const _Float16* arow = hy + (m0 + (lane & 15)) * DM;
  const _Float16* Wb   = Wp + s0 * DM;
  v8f acc = {};
  for (int k0 = 0; k0 < DM; k0 += 32) {
    acc = wmma16(load_a_row(arow, k0, lane), load_b_frag(Wb, DM, k0, lane), acc);
  }
  int s  = s0 + (lane & 15);
  int mb = m0 + ((lane & 16) ? 8 : 0);
  float bias = pb[s];
#pragma unroll
  for (int v = 0; v < 8; ++v) {
    int r = mb + v;                 // r = b*512 + c*8 + p
    int b = r >> 9;
    int c = (r >> 3) & 63;
    int p = r & 7;
    float last = x[(b * SEQN + (SEQN - 1)) * ENCN + c];
    out[b * (PREDN * ENCN) + (p * SEGL + s) * ENCN + c] = acc[v] + bias + last;
  }
}

// ---------------- host launch ----------------
extern "C" void kernel_launch(void* const* d_in, const int* in_sizes, int n_in,
                              void* d_out, int out_size, void* d_ws, size_t ws_size,
                              hipStream_t stream) {
  (void)in_sizes; (void)n_in; (void)out_size; (void)ws_size;
  const float* x     = (const float*)d_in[0];
  const float* W_emb = (const float*)d_in[1];
  const float* b_emb = (const float*)d_in[2];
  const float* cWih  = (const float*)d_in[3];
  const float* cWhh  = (const float*)d_in[4];
  const float* cbih  = (const float*)d_in[5];
  const float* cbhh  = (const float*)d_in[6];
  const float* gWih  = (const float*)d_in[7];
  const float* gWhh  = (const float*)d_in[8];
  const float* gbih  = (const float*)d_in[9];
  const float* gbhh  = (const float*)d_in[10];
  const float* resW  = (const float*)d_in[11];
  const float* resb  = (const float*)d_in[12];
  const float* pos   = (const float*)d_in[13];
  const float* chan  = (const float*)d_in[14];
  const float* predW = (const float*)d_in[15];
  const float* predb = (const float*)d_in[16];
  float* out = (float*)d_out;

  char* ws = (char*)d_ws;
  size_t off = 0;
  auto carve = [&](size_t bytes) -> char* {
    char* p = ws + off;
    off += (bytes + 255) & ~(size_t)255;
    return p;
  };
  _Float16* Wemb16  = (_Float16*)carve((size_t)DM * SEGL * 2);
  _Float16* cWih16  = (_Float16*)carve((size_t)D3 * DM * 2);
  _Float16* cWhh16  = (_Float16*)carve((size_t)D3 * DM * 2);
  _Float16* gWih16  = (_Float16*)carve((size_t)D3 * DM * 2);
  _Float16* gWhh16  = (_Float16*)carve((size_t)D3 * DM * 2);
  _Float16* resW16  = (_Float16*)carve((size_t)DM * DM * 2);
  _Float16* predW16 = (_Float16*)carve((size_t)SEGL * DM * 2);
  _Float16* pe16    = (_Float16*)carve((size_t)512 * DM * 2);
  _Float16* xt16    = (_Float16*)carve((size_t)NB * SEGL * 2);
  float*    e32     = (float*)   carve((size_t)NB * DM * 4);
  _Float16* e16     = (_Float16*)carve((size_t)NB * DM * 2);
  float*    h32     = (float*)   carve((size_t)NB * DM * 4);
  _Float16* h16     = (_Float16*)carve((size_t)NB * DM * 2);
  _Float16* hc16    = (_Float16*)carve((size_t)NB * DM * 2);
  _Float16* hy16    = (_Float16*)carve((size_t)RD * DM * 2);

  auto cvt = [&](const float* src, _Float16* dst, int n) {
    k_convert<<<(n + 255) / 256, 256, 0, stream>>>(src, dst, n);
  };
  cvt(W_emb, Wemb16, DM * SEGL);
  cvt(cWih, cWih16, D3 * DM);
  cvt(cWhh, cWhh16, D3 * DM);
  cvt(gWih, gWih16, D3 * DM);
  cvt(gWhh, gWhh16, D3 * DM);
  cvt(resW, resW16, DM * DM);
  cvt(predW, predW16, SEGL * DM);

  k_zero_h<<<NB * DM / 256, 256, 0, stream>>>(h32, h16);
  k_pe<<<512 * DM / 256, 256, 0, stream>>>(pos, chan, pe16);

  // Encoder: blocks = (Mtiles/8) * Ntiles
  const int blocks_emb   = (NB / 16 / 8) * (DM / 16);   // 64*32  = 2048
  const int blocks_gates = (NB / 32 / 8) * (DM / 16);   // 32*32  = 1024
  const int blocks_res   = (NB / 16 / 8) * (DM / 16);   // 2048
  for (int t = 0; t < SNX; ++t) {
    k_xt<<<NB * SEGL / 256, 256, 0, stream>>>(x, xt16, t);
    k_emb<<<blocks_emb, 256, 0, stream>>>(xt16, Wemb16, b_emb, e32, e16);
    k_gates<<<blocks_gates, 256, 0, stream>>>(e16, h16, h32, cWih16, cWhh16,
                                              cbih, cbhh, hc16);
    k_res<<<blocks_res, 256, 0, stream>>>(hc16, resW16, resb, e32, h32, h16);
  }

  const int blocks_dec  = (RD / 32 / 8) * (DM / 16);    // 256*32 = 8192
  k_dec<<<blocks_dec, 256, 0, stream>>>(pe16, h16, h32, gWih16, gWhh16,
                                        gbih, gbhh, hy16);
  const int blocks_pred = (RD / 16 / 8) * (SEGL / 16);  // 512*4  = 2048
  k_pred<<<blocks_pred, 256, 0, stream>>>(hy16, predW16, predb, x, out);
}
